// CommonFeatureExtractor_29678224016215
// MI455X (gfx1250) — matrix operand order repaired and checked by reading, not compiled, not run on hardware
//
#include <hip/hip_runtime.h>
#include <stdint.h>

// ---------------------------------------------------------------------------
// CDNA5 (gfx1250) bf16 WMMA implementation of CommonFeatureExtractor.
// All GEMMs use v_wmma_f32_16x16x32_bf16 (wave32), f32 accumulate.
// A-operands are pre-converted to bf16; the GEMM inner loop double-buffers
// LDS tiles filled by GLOBAL_LOAD_ASYNC_TO_LDS_B128 (ASYNCcnt), so the next
// tile's HBM fetch overlaps the current tile's WMMA work.
// ---------------------------------------------------------------------------

typedef __attribute__((ext_vector_type(16))) __bf16 bf16x16;
typedef __attribute__((ext_vector_type(8)))  __bf16 bf16x8;
typedef __attribute__((ext_vector_type(8)))  float  f32x8;
typedef __attribute__((ext_vector_type(4)))  int    v4i;

union FragBF {
    bf16x16 v;
    bf16x8  h[2];
};

__device__ __forceinline__ unsigned short f2bf(float f) {
    __bf16 h = (__bf16)f;  // native cvt (v_cvt_pk_bf16_f32 capable)
    return __builtin_bit_cast(unsigned short, h);
}

static constexpr int ACT_NONE   = 0;
static constexpr int ACT_RELU   = 1;
static constexpr int ACT_SIGMUL = 2;  // out = mulSrc * sigmoid(acc + bias)

// ---- gfx1250 async global->LDS path (ASYNCcnt) ----------------------------
#define AS_GLOBAL __attribute__((address_space(1)))
#define AS_LDS    __attribute__((address_space(3)))

#if __has_builtin(__builtin_amdgcn_global_load_async_to_lds_b128)
#define HAVE_ASYNC_LDS 1
__device__ __forceinline__ void async_copy_b128(const void* gsrc, void* ldst) {
    // Builtin signature: (global v4i*, lds v4i*, imm offset, imm cpol).
    // generic->AS casts via integer: low 32 bits of a generic LDS pointer are
    // the LDS byte offset (flat aperture rule), AS1 keeps the full 64-bit VA.
    __builtin_amdgcn_global_load_async_to_lds_b128(
        (AS_GLOBAL v4i*)(uintptr_t)gsrc, (AS_LDS v4i*)(uintptr_t)ldst, 0, 0);
}
__device__ __forceinline__ void wait_async0() {
#if __has_builtin(__builtin_amdgcn_s_wait_asynccnt)
    __builtin_amdgcn_s_wait_asynccnt(0);
#else
    asm volatile("s_wait_asynccnt 0x0" ::: "memory");
#endif
}
#else
#define HAVE_ASYNC_LDS 0
#endif

// ---------------------------------------------------------------------------
// Activation convert: f32 -> bf16, row-major, 4 elems/thread.
// ---------------------------------------------------------------------------
__global__ void convert_act_kernel(const float* __restrict__ src,
                                   unsigned short* __restrict__ dst,
                                   int total4) {
    int i = blockIdx.x * 256 + threadIdx.x;
    if (i < total4) {
        float4 v = *(const float4*)(src + (size_t)i * 4);
        uint2 p;
        p.x = (unsigned)f2bf(v.x) | ((unsigned)f2bf(v.y) << 16);
        p.y = (unsigned)f2bf(v.z) | ((unsigned)f2bf(v.w) << 16);
        *(uint2*)(dst + (size_t)i * 4) = p;
    }
}

// ---------------------------------------------------------------------------
// Weight convert + transpose via 32x32 LDS tile (both global sides coalesced):
// W (K x N f32 row-major) -> WT (Npad x Kpad bf16 row-major, zero padded).
// Grid: (Kpad/32, Npad/32), 256 threads (32x8).
// ---------------------------------------------------------------------------
__global__ void convert_wT_kernel(const float* __restrict__ W,
                                  unsigned short* __restrict__ WT,
                                  int K, int N, int Kpad, int Npad) {
    __shared__ unsigned short sT[32][33];
    const int tx = threadIdx.x & 31;
    const int ty = threadIdx.x >> 5;  // 0..7
    const int kBase = blockIdx.x * 32;
    const int nBase = blockIdx.y * 32;
#pragma unroll
    for (int i = 0; i < 4; ++i) {
        int k = kBase + ty + i * 8;
        int n = nBase + tx;
        float v = (k < K && n < N) ? W[(size_t)k * N + n] : 0.0f;
        sT[ty + i * 8][tx] = f2bf(v);
    }
    __syncthreads();
#pragma unroll
    for (int i = 0; i < 4; ++i) {
        int n = nBase + ty + i * 8;
        int k = kBase + tx;
        WT[(size_t)n * Kpad + k] = sT[tx][ty + i * 8];
    }
}

// ---------------------------------------------------------------------------
// Tiled WMMA GEMM:  out = act( A(MxK bf16) @ BT^T + bias )
//   BT : bf16, (>= gridDim.x*128) x ldbt row-major (pre-transposed, padded)
// Block tile 128x128, 256 threads = 8 waves, wave tile 64x32 (4x2 WMMAs).
// Double-buffered LDS: one barrier per K-step, fetch of tile i+1 overlaps
// WMMA of tile i. M % 128 == 0, K % 32 == 0; B-tiles in bounds (padding).
// ---------------------------------------------------------------------------
template <int ACT, bool OUTBF16>
__global__ __launch_bounds__(256) void gemm_wmma_kernel(
    const unsigned short* __restrict__ A, int lda,
    const unsigned short* __restrict__ BT, int ldbt,
    const float* __restrict__ bias, int Nbias,
    void* __restrict__ outPtr, int ldOut, int Nout,
    const float* __restrict__ mulSrc, int ldMul,
    int K) {
    // pitch 40 bf16 (80 B): 16-B aligned chunks, no bank-conflict periodicity
    constexpr int TILE = 128 * 40;  // ushorts per tile buffer
    __shared__ __attribute__((aligned(16))) unsigned short sA[2 * TILE];
    __shared__ __attribute__((aligned(16))) unsigned short sB[2 * TILE];

    const int tid   = threadIdx.x;
    const int lane  = tid & 31;
    const int wave  = tid >> 5;
    const int r     = lane & 15;   // row (A) / col (B,C) within 16x16
    const int hi    = lane >> 4;   // half-wave select (K grouping)
    const int waveM = wave & 1;    // 2 waves along M
    const int waveN = wave >> 1;   // 4 waves along N
    const int mBase = blockIdx.y * 128;
    const int nBase = blockIdx.x * 128;

    // per-thread tile chunk coordinates (512 chunks of 8 bf16 per tile)
    const int c0row = tid >> 2, c0col = (tid & 3) * 8;
    const int c1row = (tid + 256) >> 2, c1col = ((tid + 256) & 3) * 8;

#if HAVE_ASYNC_LDS
    auto issue_tile = [&](int kt, int buf) {
        const unsigned short* gA = A + (size_t)mBase * lda + kt;
        const unsigned short* gB = BT + (size_t)nBase * ldbt + kt;
        unsigned short* dA = &sA[buf * TILE];
        unsigned short* dB = &sB[buf * TILE];
        async_copy_b128(gA + (size_t)c0row * lda + c0col, dA + c0row * 40 + c0col);
        async_copy_b128(gA + (size_t)c1row * lda + c1col, dA + c1row * 40 + c1col);
        async_copy_b128(gB + (size_t)c0row * ldbt + c0col, dB + c0row * 40 + c0col);
        async_copy_b128(gB + (size_t)c1row * ldbt + c1col, dB + c1row * 40 + c1col);
    };
#endif

    f32x8 acc[4][2];
#pragma unroll
    for (int mt = 0; mt < 4; ++mt)
#pragma unroll
        for (int nt = 0; nt < 2; ++nt)
            acc[mt][nt] = (f32x8){0.f, 0.f, 0.f, 0.f, 0.f, 0.f, 0.f, 0.f};

#if HAVE_ASYNC_LDS
    issue_tile(0, 0);
#else
    uint4 ra0, ra1, rb0, rb1;
    {
        const unsigned short* gA = A + (size_t)mBase * lda;
        const unsigned short* gB = BT + (size_t)nBase * ldbt;
        ra0 = *(const uint4*)(gA + (size_t)c0row * lda + c0col);
        ra1 = *(const uint4*)(gA + (size_t)c1row * lda + c1col);
        rb0 = *(const uint4*)(gB + (size_t)c0row * ldbt + c0col);
        rb1 = *(const uint4*)(gB + (size_t)c1row * ldbt + c1col);
    }
#endif

    int cur = 0;
    for (int kt = 0; kt < K; kt += 32) {
        const unsigned short* cA = &sA[cur * TILE];
        const unsigned short* cB = &sB[cur * TILE];
#if HAVE_ASYNC_LDS
        wait_async0();       // tile kt (this wave's asyncs) landed in LDS
        __syncthreads();     // all waves landed; prior reads of other buf done
        if (kt + 32 < K) issue_tile(kt + 32, cur ^ 1);  // overlap with WMMA
#else
        {
            unsigned short* dA = &sA[cur * TILE];
            unsigned short* dB = &sB[cur * TILE];
            *(uint4*)(dA + c0row * 40 + c0col) = ra0;
            *(uint4*)(dA + c1row * 40 + c1col) = ra1;
            *(uint4*)(dB + c0row * 40 + c0col) = rb0;
            *(uint4*)(dB + c1row * 40 + c1col) = rb1;
        }
        __syncthreads();
        if (kt + 32 < K) {   // prefetch next tile into registers
            const unsigned short* gA = A + (size_t)mBase * lda + kt + 32;
            const unsigned short* gB = BT + (size_t)nBase * ldbt + kt + 32;
            ra0 = *(const uint4*)(gA + (size_t)c0row * lda + c0col);
            ra1 = *(const uint4*)(gA + (size_t)c1row * lda + c1col);
            rb0 = *(const uint4*)(gB + (size_t)c0row * ldbt + c0col);
            rb1 = *(const uint4*)(gB + (size_t)c1row * ldbt + c1col);
        }
#endif

        // ---- fragments per ISA layout (CDNA5 ISA §7.12.2) ----
        // A 16x32 bf16: lane r (hi=0): K 0..7 & 16..23 ; hi=1: K 8..15 & 24..31
        FragBF aF[4];
#pragma unroll
        for (int mt = 0; mt < 4; ++mt) {
            int aRow = waveM * 64 + mt * 16 + r;
            aF[mt].h[0] = *(const bf16x8*)(cA + aRow * 40 + hi * 8);
            aF[mt].h[1] = *(const bf16x8*)(cA + aRow * 40 + 16 + hi * 8);
        }
        // B 32x16 bf16: lane r = col, hi=0: K 0..7 & 16..23 ; hi=1: K 8..15 & 24..31
        FragBF bF[2];
#pragma unroll
        for (int nt = 0; nt < 2; ++nt) {
            int bRow = waveN * 32 + nt * 16 + r;
            bF[nt].h[0] = *(const bf16x8*)(cB + bRow * 40 + hi * 8);
            bF[nt].h[1] = *(const bf16x8*)(cB + bRow * 40 + 16 + hi * 8);
        }
#pragma unroll
        for (int mt = 0; mt < 4; ++mt)
#pragma unroll
            for (int nt = 0; nt < 2; ++nt)
                acc[mt][nt] = __builtin_amdgcn_wmma_f32_16x16x32_bf16(
                    false, aF[mt].v, false, bF[nt].v, (short)0, acc[mt][nt],
                    false, false);
        cur ^= 1;
    }

    // ---- epilogue: C layout -> element (m = hi*8+i, n = lane&15) ----
#pragma unroll
    for (int mt = 0; mt < 4; ++mt) {
#pragma unroll
        for (int nt = 0; nt < 2; ++nt) {
            int n = nBase + waveN * 32 + nt * 16 + r;
            if (n < Nout) {
                float bv = (n < Nbias) ? bias[n] : 0.0f;
#pragma unroll
                for (int i = 0; i < 8; ++i) {
                    int m   = mBase + waveM * 64 + mt * 16 + hi * 8 + i;
                    float v = acc[mt][nt][i] + bv;
                    if constexpr (ACT == ACT_RELU) v = fmaxf(v, 0.0f);
                    if constexpr (ACT == ACT_SIGMUL) {
                        float sg = 1.0f / (1.0f + __expf(-v));
                        v = mulSrc[(size_t)m * ldMul + n] * sg;
                    }
                    if constexpr (OUTBF16)
                        ((unsigned short*)outPtr)[(size_t)m * ldOut + n] = f2bf(v);
                    else
                        ((float*)outPtr)[(size_t)m * ldOut + n] = v;
                }
            }
        }
    }
}

// ---------------------------------------------------------------------------
// Row-wise fusion: norms, pairwise sims, keep/elem masks, both 3-way
// softmaxes, common feature and weighted_fp_sum. One block per row (H=1024).
// Writes commonF (f32), commonBf (bf16 GEMM operand), fusedIn[:, 0:1024].
// ---------------------------------------------------------------------------
__global__ __launch_bounds__(256) void rowfuse_kernel(
    const float* __restrict__ E0, const float* __restrict__ E1,
    const float* __restrict__ E2, const float* __restrict__ wgW,
    const float* __restrict__ wgb, float* __restrict__ commonF,
    unsigned short* __restrict__ commonBf,
    unsigned short* __restrict__ fusedIn) {
    const int H   = 1024;
    const int b   = blockIdx.x;
    const int tid = threadIdx.x;
    const float* eb = E0 + (size_t)b * H;
    const float* ef = E1 + (size_t)b * H;
    const float* ep = E2 + (size_t)b * H;

    float vb[4], vf[4], vp[4];
    float part[9];
#pragma unroll
    for (int c = 0; c < 9; ++c) part[c] = 0.0f;

#pragma unroll
    for (int it = 0; it < 4; ++it) {
        int h   = tid + it * 256;
        float a = eb[h], f = ef[h], p = ep[h];
        vb[it] = a; vf[it] = f; vp[it] = p;
        part[0] += a * a; part[1] += f * f; part[2] += p * p;
        part[3] += a * f; part[4] += a * p; part[5] += f * p;
        const float* w0r = wgW + (size_t)h * 3;
        const float* w1r = wgW + (size_t)(H + h) * 3;
        const float* w2r = wgW + (size_t)(2 * H + h) * 3;
        part[6] += a * w0r[0] + f * w1r[0] + p * w2r[0];
        part[7] += a * w0r[1] + f * w1r[1] + p * w2r[1];
        part[8] += a * w0r[2] + f * w1r[2] + p * w2r[2];
    }

    __shared__ float red[9][256];
#pragma unroll
    for (int c = 0; c < 9; ++c) red[c][tid] = part[c];
    for (int s = 128; s > 0; s >>= 1) {
        __syncthreads();
        if (tid < s)
#pragma unroll
            for (int c = 0; c < 9; ++c) red[c][tid] += red[c][tid + s];
    }
    __syncthreads();

    float nb  = fmaxf(sqrtf(red[0][0]), 1e-12f);
    float nf  = fmaxf(sqrtf(red[1][0]), 1e-12f);
    float npn = fmaxf(sqrtf(red[2][0]), 1e-12f);
    float sim0 = red[3][0] / (nb * nf);
    float sim1 = red[4][0] / (nb * npn);
    float sim2 = red[5][0] / (nf * npn);
    bool k0 = sim0 > 0.6f, k1 = sim1 > 0.6f, k2 = sim2 > 0.6f;
    float s0 = k0 ? sim0 : -1e9f;
    float s1 = k1 ? sim1 : -1e9f;
    float s2 = k2 ? sim2 : -1e9f;
    float sm  = fmaxf(s0, fmaxf(s1, s2));
    float e0 = __expf(s0 - sm), e1 = __expf(s1 - sm), e2 = __expf(s2 - sm);
    float ei = 1.0f / (e0 + e1 + e2);
    float w0 = e0 * ei, w1 = e1 * ei, w2 = e2 * ei;

    float g0 = red[6][0] + wgb[0];
    float g1 = red[7][0] + wgb[1];
    float g2 = red[8][0] + wgb[2];
    float gm = fmaxf(g0, fmaxf(g1, g2));
    float x0 = __expf(g0 - gm), x1 = __expf(g1 - gm), x2 = __expf(g2 - gm);
    float xi = 1.0f / (x0 + x1 + x2);
    float f0 = x0 * xi, f1 = x1 * xi, f2 = x2 * xi;

    bool anyk = k0 || k1 || k2;
    float rb = 1.0f / nb, rf = 1.0f / nf, rp = 1.0f / npn;

#pragma unroll
    for (int it = 0; it < 4; ++it) {
        int h   = tid + it * 256;
        float a = vb[it], f = vf[it], p = vp[it];
        float na = a * rb, nfv = f * rf, npv = p * rp;
        float c0 = ((na * nfv)  > 0.6f) ? 0.5f * (a + f) : 0.0f;
        float c1 = ((na * npv)  > 0.6f) ? 0.5f * (a + p) : 0.0f;
        float c2 = ((nfv * npv) > 0.6f) ? 0.5f * (f + p) : 0.0f;
        float wsum = w0 * c0 + w1 * c1 + w2 * c2;
        float mean = (a + f + p) * (1.0f / 3.0f);
        float common = anyk ? wsum : mean;
        commonF[(size_t)b * H + h] = common;
        commonBf[(size_t)b * H + h] = f2bf(common);
        float wfp = f0 * a + f1 * f + f2 * p;
        fusedIn[(size_t)b * 2048 + h] = f2bf(wfp);
    }
}

// ---------------------------------------------------------------------------
extern "C" void kernel_launch(void* const* d_in, const int* in_sizes, int n_in,
                              void* d_out, int out_size, void* d_ws,
                              size_t ws_size, hipStream_t stream) {
    (void)in_sizes; (void)n_in; (void)out_size; (void)ws_size;
    const int B = 16384, H = 1024, D = 300, Dp = 320, DpB = 384, H2 = 2048;

    const float* X[3]  = {(const float*)d_in[0], (const float*)d_in[1], (const float*)d_in[2]};
    const float* W1[3] = {(const float*)d_in[3], (const float*)d_in[7],  (const float*)d_in[11]};
    const float* B1[3] = {(const float*)d_in[4], (const float*)d_in[8],  (const float*)d_in[12]};
    const float* W2[3] = {(const float*)d_in[5], (const float*)d_in[9],  (const float*)d_in[13]};
    const float* B2[3] = {(const float*)d_in[6], (const float*)d_in[10], (const float*)d_in[14]};
    const float* wgW   = (const float*)d_in[15];
    const float* wgb   = (const float*)d_in[16];
    const float* enhW  = (const float*)d_in[17];
    const float* enhb  = (const float*)d_in[18];
    const float* fusW  = (const float*)d_in[19];
    const float* fusb  = (const float*)d_in[20];

    char* ws = (char*)d_ws;
    size_t off = 0;
    auto take = [&](size_t nbytes) -> char* {
        char* p = ws + off;
        off += (nbytes + 255) & ~(size_t)255;
        return p;
    };

    unsigned short* w1T[3];   // DpB x H  (padded to 384 rows: branch-free B loads)
    unsigned short* w2T[3];   // H x Dp
    for (int i = 0; i < 3; ++i) w1T[i] = (unsigned short*)take((size_t)DpB * H * 2);
    for (int i = 0; i < 3; ++i) w2T[i] = (unsigned short*)take((size_t)H * Dp * 2);
    unsigned short* enhT     = (unsigned short*)take((size_t)H * H * 2);
    unsigned short* fusT     = (unsigned short*)take((size_t)H * H2 * 2);
    unsigned short* Xbf      = (unsigned short*)take((size_t)B * H * 2);   // reused per MLP
    unsigned short* Hbf      = (unsigned short*)take((size_t)B * Dp * 2);  // reused per MLP
    float*          E[3];
    for (int i = 0; i < 3; ++i) E[i] = (float*)take((size_t)B * H * 4);
    float*          commonF  = (float*)take((size_t)B * H * 4);
    unsigned short* commonBf = (unsigned short*)take((size_t)B * H * 2);
    unsigned short* fusedIn  = (unsigned short*)take((size_t)B * H2 * 2);

    dim3 blk(256);
    auto cgrid = [](int total) { return dim3((unsigned)((total + 255) / 256)); };

    // ---- weight convert + transpose (bf16, zero-padded) ----
    for (int i = 0; i < 3; ++i)
        convert_wT_kernel<<<dim3(H / 32, DpB / 32), blk, 0, stream>>>(
            W1[i], w1T[i], H, D, H, DpB);
    for (int i = 0; i < 3; ++i)
        convert_wT_kernel<<<dim3(Dp / 32, H / 32), blk, 0, stream>>>(
            W2[i], w2T[i], D, H, Dp, H);
    convert_wT_kernel<<<dim3(H / 32,  H / 32), blk, 0, stream>>>(enhW, enhT, H,  H, H,  H);
    convert_wT_kernel<<<dim3(H2 / 32, H / 32), blk, 0, stream>>>(fusW, fusT, H2, H, H2, H);

    const int mTiles = B / 128;  // 128

    // ---- three MLPs: relu(X@W1+b1) -> Hbf (bf16, padded 320), @W2+b2 -> E ----
    for (int i = 0; i < 3; ++i) {
        convert_act_kernel<<<cgrid(B * H / 4), blk, 0, stream>>>(X[i], Xbf, B * H / 4);
        gemm_wmma_kernel<ACT_RELU, true>
            <<<dim3(Dp / 128 + 1, mTiles), blk, 0, stream>>>(  // 3 tiles cover 384
                Xbf, H, w1T[i], H, B1[i], D,
                Hbf, Dp, Dp, nullptr, 0, H);
        gemm_wmma_kernel<ACT_NONE, false>
            <<<dim3(H / 128, mTiles), blk, 0, stream>>>(
                Hbf, Dp, w2T[i], Dp, B2[i], H,
                E[i], H, H, nullptr, 0, Dp);
    }

    // ---- row-wise fusion: common (f32+bf16) + weighted_fp_sum (fusedIn[:, :H]) ----
    rowfuse_kernel<<<dim3(B), blk, 0, stream>>>(E[0], E[1], E[2], wgW, wgb,
                                                commonF, commonBf, fusedIn);

    // ---- enhanced_common = common * sigmoid(common@enhW + b) -> fusedIn[:, H:] ----
    gemm_wmma_kernel<ACT_SIGMUL, true>
        <<<dim3(H / 128, mTiles), blk, 0, stream>>>(
            commonBf, H, enhT, H, enhb, H,
            (void*)(fusedIn + H), H2, H, commonF, H, H);

    // ---- fused = [weighted_fp_sum, enhanced_common] @ fusW + b -> d_out ----
    gemm_wmma_kernel<ACT_NONE, false>
        <<<dim3(H / 128, mTiles), blk, 0, stream>>>(
            fusedIn, H2, fusT, H2, fusb, H,
            d_out, H, H, nullptr, 0, H2);
}
